// AttentionMachineTranslationModel_76854144795158
// MI455X (gfx1250) — matrix operand order repaired and checked by reading, not compiled, not run on hardware
//
#include <hip/hip_runtime.h>
#include <hip/hip_bf16.h>
#include <math.h>

#define DEVFN __device__ __forceinline__

typedef __attribute__((ext_vector_type(16))) __bf16 v16bf;
typedef __attribute__((ext_vector_type(8)))  float  v8f;

// Model dims
#define VV 32000
#define EE 256
#define UU 512
#define DD 512
#define BB 32
#define TT 64
#define G4 2048   // 4*U == 4*D

// ---------- helpers ----------
DEVFN __bf16 f2bf(float f) {
    unsigned u = __builtin_bit_cast(unsigned, f);
    unsigned short s = (unsigned short)((u + 0x7FFFu + ((u >> 16) & 1u)) >> 16);
    return __builtin_bit_cast(__bf16, s);
}

DEVFN float sigm(float x) { return 1.0f / (1.0f + __expf(-x)); }

DEVFN v8f wmma_bf16(v16bf a, v16bf b, v8f c) {
    // D = A(16x32) * B(32x16) + C, f32 accumulate
    return __builtin_amdgcn_wmma_f32_16x16x32_bf16(false, a, false, b, (short)0, c, false, false);
}

// Packed fragment load: each lane reads its 16 bf16 (32 bytes, 32B-aligned)
// -> two coalesced b128 loads per wave-fragment.
DEVFN v16bf loadFrag(const __bf16* base) {
    return *(const v16bf*)(base + (size_t)(threadIdx.x & 31) * 16);
}

// Inverse A-fragment index: element (row b, col u) of a matrix stored
// fragment-major [mt][kt(ktiles)][lane(32)][elem(16)].
// A 16x32 bf16 layout: lane = half*16 + m; VGPR j holds K = 2j + (j>=4?8:0) + 8*half (+1 hi).
DEVFN int fragAoff(int ktiles, int b, int u) {
    int mt = b >> 4, m = b & 15;
    int kt = u >> 5, kl = u & 31;
    int half = (kl >> 3) & 1;
    int base = kl - 8 * half;                 // in [0,8) u [16,24)
    int j = (base < 8) ? (base >> 1) : (((base - 16) >> 1) + 4);
    int i = 2 * j + (base & 1);
    return ((mt * ktiles + kt) * 32 + (half * 16 + m)) * 16 + i;
}

// ---------- kernels ----------

// Repack f32 row-major W[K][N] into bf16 B-fragments [nt][kt][lane][16].
// B 32x16 layout: n = lane&15; VGPR j holds K = 2j + (j>=4?8:0) + 8*(lane>>4) (+1 hi).
__global__ void pack_w(const float* __restrict__ W, __bf16* __restrict__ P, int K, int N) {
    int idx = blockIdx.x * blockDim.x + threadIdx.x;
    if (idx >= K * N) return;
    int i    = idx & 15;
    int lane = (idx >> 4) & 31;
    int frag = idx >> 9;
    int ktiles = K >> 5;
    int nt = frag / ktiles, kt = frag - nt * ktiles;
    int half = lane >> 4, n = lane & 15;
    int j = i >> 1;
    int kl = 2 * j + (j >= 4 ? 8 : 0) + 8 * half + (i & 1);
    P[idx] = f2bf(W[(size_t)(kt * 32 + kl) * N + (nt * 16 + n)]);
}

// Embedding gather, written directly in A-fragment layout:
// xs_p[t][mt(2)][kt(8)][lane(32)][16]  (8192 elems per t)
__global__ void embed_k(const int* __restrict__ x, const float* __restrict__ emb,
                        __bf16* __restrict__ xsp) {
    int idx = blockIdx.x * blockDim.x + threadIdx.x;
    if (idx >= TT * BB * EE) return;
    int i    = idx & 15;
    int lane = (idx >> 4) & 31;
    int kt   = (idx >> 9) & 7;
    int mt   = (idx >> 12) & 1;
    int t    = idx >> 13;
    int half = lane >> 4, m = lane & 15;
    int j = i >> 1;
    int kb = 2 * j + (j >= 4 ? 8 : 0) + 8 * half + (i & 1);
    int b = mt * 16 + m;
    int e = kt * 32 + kb;
    int tok = x[b * TT + t];
    xsp[idx] = f2bf(emb[(size_t)tok * EE + e]);
}

// Bidirectional encoder LSTM: blockIdx.x = direction, 512 threads = 16 waves.
__global__ __launch_bounds__(512)
void encoder_k(const __bf16* __restrict__ xsp,
               const __bf16* __restrict__ WxF, const __bf16* __restrict__ WhF, const float* __restrict__ bF,
               const __bf16* __restrict__ WxB, const __bf16* __restrict__ WhB, const float* __restrict__ bB,
               float* __restrict__ a) {
    const int dir = blockIdx.x;
    const __bf16* Wx = dir ? WxB : WxF;
    const __bf16* Wh = dir ? WhB : WhF;
    const float*  bi = dir ? bB  : bF;

    // current hidden state, bf16, A-fragment layout [mt(2)][kt(16)][lane(32)][16] = 32 KB
    __shared__ __align__(32) __bf16 hsh[BB * UU];
    for (int i = threadIdx.x; i < BB * UU; i += 512) hsh[i] = f2bf(0.0f);

    const int wave = threadIdx.x >> 5;
    const int lane = threadIdx.x & 31;
    const int half = lane >> 4, n16 = lane & 15;

    v8f c[4];
#pragma unroll
    for (int q = 0; q < 4; ++q)
#pragma unroll
        for (int e = 0; e < 8; ++e) c[q][e] = 0.0f;

    __syncthreads();

    for (int t = 0; t < TT; ++t) {
        const int ts = dir ? (TT - 1 - t) : t;
        const __bf16* xt = xsp + (size_t)ts * (BB * EE);
        v8f hsv[4];

        for (int q = 0; q < 4; ++q) {
            const int p = wave + 16 * q;
            const int mt = p & 1;
            const int ut = p >> 1;

            v8f acc[4];
#pragma unroll
            for (int g = 0; g < 4; ++g) {
                float bv = bi[g * UU + ut * 16 + n16];
#pragma unroll
                for (int e = 0; e < 8; ++e) acc[g][e] = bv;
            }
            // x_t @ Wx   (K = 256, 8 k-tiles)
            for (int kk = 0; kk < 8; ++kk) {
                v16bf af = loadFrag(xt + (size_t)(mt * 8 + kk) * 512);
#pragma unroll
                for (int g = 0; g < 4; ++g) {
                    v16bf bf_ = loadFrag(Wx + (size_t)((g * 32 + ut) * 8 + kk) * 512);
                    acc[g] = wmma_bf16(af, bf_, acc[g]);
                }
            }
            // h @ Wh     (K = 512, 16 k-tiles)
            for (int kk = 0; kk < 16; ++kk) {
                v16bf af = loadFrag((const __bf16*)hsh + (size_t)(mt * 16 + kk) * 512);
#pragma unroll
                for (int g = 0; g < 4; ++g) {
                    v16bf bf_ = loadFrag(Wh + (size_t)((g * 32 + ut) * 16 + kk) * 512);
                    acc[g] = wmma_bf16(af, bf_, acc[g]);
                }
            }
            // gates: i, f, g, o
#pragma unroll
            for (int e = 0; e < 8; ++e) {
                float ig = sigm(acc[0][e]);
                float fg = sigm(acc[1][e]);
                float gg = tanhf(acc[2][e]);
                float og = sigm(acc[3][e]);
                float cn = fg * c[q][e] + ig * gg;
                float hv = og * tanhf(cn);
                c[q][e] = cn;
                hsv[q][e] = hv;
                int bRow = mt * 16 + e + 8 * half;
                int u = ut * 16 + n16;
                a[((size_t)bRow * TT + ts) * (2 * UU) + dir * UU + u] = hv;
            }
        }
        __syncthreads();   // all reads of old h complete
#pragma unroll
        for (int q = 0; q < 4; ++q) {
            const int p = wave + 16 * q;
            const int mt = p & 1, ut = p >> 1;
#pragma unroll
            for (int e = 0; e < 8; ++e) {
                int bRow = mt * 16 + e + 8 * half;
                int u = ut * 16 + n16;
                hsh[fragAoff(16, bRow, u)] = f2bf(hsv[q][e]);
            }
        }
        __syncthreads();   // new h visible
    }
}

// aW1[b][t][j] = a[b][t][:] . W1[0:1024][j]   (decoder-step invariant, hoisted)
__global__ void aW1_k(const float* __restrict__ a, const float* __restrict__ W1,
                      float* __restrict__ aW1) {
    int i = blockIdx.x * blockDim.x + threadIdx.x;
    if (i >= BB * TT) return;
    const float* row = a + (size_t)i * (2 * UU);
    float s[10];
#pragma unroll
    for (int j = 0; j < 10; ++j) s[j] = 0.0f;
    for (int k = 0; k < 2 * UU; ++k) {
        float av = row[k];
        const float* w = W1 + (size_t)k * 10;
#pragma unroll
        for (int j = 0; j < 10; ++j) s[j] += av * w[j];
    }
#pragma unroll
    for (int j = 0; j < 10; ++j) aW1[(size_t)i * 10 + j] = s[j];
}

__global__ void dec_init(float* __restrict__ h0f, float* __restrict__ c0,
                         __bf16* __restrict__ h0b) {
    int i = blockIdx.x * blockDim.x + threadIdx.x;
    if (i < BB * DD) { h0f[i] = 0.0f; c0[i] = 0.0f; h0b[i] = f2bf(0.0f); }
}

// Attention for one decoder step: grid = B, block = T threads.
// ctx written in A-fragment layout (ktiles = 32) for the decoder cell GEMM.
__global__ void attn_k(const float* __restrict__ a, const float* __restrict__ aW1,
                       const float* __restrict__ W1, const float* __restrict__ b1,
                       const float* __restrict__ W2, const float* __restrict__ b2,
                       const float* __restrict__ hdec, __bf16* __restrict__ ctxp, int t) {
    (void)t;
    const int b = blockIdx.x, tid = threadIdx.x;
    __shared__ float hW1[10];
    __shared__ float sc[TT];
    __shared__ float red[TT];

    if (tid < 10) {
        float s = 0.0f;
        for (int k = 0; k < DD; ++k) s += hdec[b * DD + k] * W1[(size_t)(2 * UU + k) * 10 + tid];
        hW1[tid] = s;
    }
    __syncthreads();

    float s = 0.0f;
#pragma unroll
    for (int j = 0; j < 10; ++j) {
        float v = tanhf(aW1[((size_t)b * TT + tid) * 10 + j] + hW1[j] + b1[j]);
        s += v * W2[j];
    }
    float e = s + b2[0];
    e = e > 0.0f ? e : 0.0f;

    red[tid] = e;
    __syncthreads();
    for (int o = TT / 2; o > 0; o >>= 1) {
        if (tid < o) red[tid] = fmaxf(red[tid], red[tid + o]);
        __syncthreads();
    }
    float mx = red[0];
    __syncthreads();
    float ex = __expf(e - mx);
    sc[tid] = ex;
    red[tid] = ex;
    __syncthreads();
    for (int o = TT / 2; o > 0; o >>= 1) {
        if (tid < o) red[tid] += red[tid + o];
        __syncthreads();
    }
    float inv = 1.0f / red[0];
    __syncthreads();

    for (int u = tid; u < 2 * UU; u += TT) {
        float acc = 0.0f;
        for (int tt = 0; tt < TT; ++tt)
            acc += sc[tt] * a[((size_t)b * TT + tt) * (2 * UU) + u];
        ctxp[fragAoff(32, b, u)] = f2bf(acc * inv);
    }
}

// Decoder LSTM cell (one step): 1 block, 512 threads = 16 waves.
__global__ __launch_bounds__(512)
void deccell_k(const __bf16* __restrict__ ctxp, const __bf16* __restrict__ hinb,
               const float* __restrict__ cin,
               const __bf16* __restrict__ Wxd, const __bf16* __restrict__ Whd,
               const float* __restrict__ bd,
               float* __restrict__ houtf, __bf16* __restrict__ houtb,
               float* __restrict__ cout) {
    const int wave = threadIdx.x >> 5;
    const int lane = threadIdx.x & 31;
    const int half = lane >> 4, n16 = lane & 15;

    for (int q = 0; q < 4; ++q) {
        const int p = wave + 16 * q;
        const int mt = p & 1, ut = p >> 1;

        v8f acc[4];
#pragma unroll
        for (int g = 0; g < 4; ++g) {
            float bv = bd[g * DD + ut * 16 + n16];
#pragma unroll
            for (int e = 0; e < 8; ++e) acc[g][e] = bv;
        }
        // ctx @ Wxd  (K = 1024, 32 k-tiles)
        for (int kk = 0; kk < 32; ++kk) {
            v16bf af = loadFrag(ctxp + (size_t)(mt * 32 + kk) * 512);
#pragma unroll
            for (int g = 0; g < 4; ++g) {
                v16bf bf_ = loadFrag(Wxd + (size_t)((g * 32 + ut) * 32 + kk) * 512);
                acc[g] = wmma_bf16(af, bf_, acc[g]);
            }
        }
        // h @ Whd    (K = 512, 16 k-tiles)
        for (int kk = 0; kk < 16; ++kk) {
            v16bf af = loadFrag(hinb + (size_t)(mt * 16 + kk) * 512);
#pragma unroll
            for (int g = 0; g < 4; ++g) {
                v16bf bf_ = loadFrag(Whd + (size_t)((g * 32 + ut) * 16 + kk) * 512);
                acc[g] = wmma_bf16(af, bf_, acc[g]);
            }
        }
#pragma unroll
        for (int e = 0; e < 8; ++e) {
            int bRow = mt * 16 + e + 8 * half;
            int u = ut * 16 + n16;
            int idx = bRow * DD + u;
            float cv = cin[idx];
            float ig = sigm(acc[0][e]);
            float fg = sigm(acc[1][e]);
            float gg = tanhf(acc[2][e]);
            float og = sigm(acc[3][e]);
            float cn = fg * cv + ig * gg;
            float hv = og * tanhf(cn);
            cout[idx]  = cn;
            houtf[idx] = hv;
            houtb[fragAoff(16, bRow, u)] = f2bf(hv);
        }
    }
}

// logits[b][v] = h . Wo[:,v] + bo[v]; 2000 n-tiles, one per wave, both m-tiles.
__global__ __launch_bounds__(256)
void outproj_k(const __bf16* __restrict__ hb, const __bf16* __restrict__ Wo,
               const float* __restrict__ bo, float* __restrict__ logits) {
    const int w = blockIdx.x * (blockDim.x >> 5) + (threadIdx.x >> 5);
    if (w >= VV / 16) return;
    const int lane = threadIdx.x & 31;
    const int half = lane >> 4, n16 = lane & 15;
    const int nt = w;

    float bv = bo[nt * 16 + n16];
    v8f acc0, acc1;
#pragma unroll
    for (int e = 0; e < 8; ++e) { acc0[e] = bv; acc1[e] = bv; }

    for (int kk = 0; kk < 16; ++kk) {
        if (kk + 2 < 16)
            __builtin_prefetch(Wo + (size_t)(nt * 16 + kk + 2) * 512, 0, 1);
        v16bf bf_ = loadFrag(Wo + (size_t)(nt * 16 + kk) * 512);
        v16bf a0  = loadFrag(hb + (size_t)kk * 512);
        v16bf a1  = loadFrag(hb + (size_t)(16 + kk) * 512);
        acc0 = wmma_bf16(a0, bf_, acc0);
        acc1 = wmma_bf16(a1, bf_, acc1);
    }
#pragma unroll
    for (int e = 0; e < 8; ++e) {
        int m = e + 8 * half;
        int v = nt * 16 + n16;
        logits[(size_t)m * VV + v]        = acc0[e];
        logits[(size_t)(16 + m) * VV + v] = acc1[e];
    }
}

// Row-wise stable softmax over V into d_out[b][t][:]
__global__ __launch_bounds__(256)
void softmax_k(const float* __restrict__ logits, float* __restrict__ out, int t) {
    const int b = blockIdx.x, tid = threadIdx.x;
    __shared__ float red[256];
    const float* row = logits + (size_t)b * VV;

    float mx = -1e30f;
    for (int v = tid; v < VV; v += 256) mx = fmaxf(mx, row[v]);
    red[tid] = mx;
    __syncthreads();
    for (int o = 128; o > 0; o >>= 1) {
        if (tid < o) red[tid] = fmaxf(red[tid], red[tid + o]);
        __syncthreads();
    }
    mx = red[0];
    __syncthreads();

    float s = 0.0f;
    for (int v = tid; v < VV; v += 256) s += __expf(row[v] - mx);
    red[tid] = s;
    __syncthreads();
    for (int o = 128; o > 0; o >>= 1) {
        if (tid < o) red[tid] += red[tid + o];
        __syncthreads();
    }
    float inv = 1.0f / red[0];

    float* o = out + ((size_t)b * TT + t) * VV;
    for (int v = tid; v < VV; v += 256) o[v] = __expf(row[v] - mx) * inv;
}

// ---------- host ----------
extern "C" void kernel_launch(void* const* d_in, const int* in_sizes, int n_in,
                              void* d_out, int out_size, void* d_ws, size_t ws_size,
                              hipStream_t stream) {
    (void)in_sizes; (void)n_in; (void)out_size; (void)ws_size;

    const int*   x    = (const int*)  d_in[0];
    const float* emb  = (const float*)d_in[1];
    const float* Wxf  = (const float*)d_in[2];
    const float* Whf  = (const float*)d_in[3];
    const float* bf   = (const float*)d_in[4];
    const float* Wxb  = (const float*)d_in[5];
    const float* Whb  = (const float*)d_in[6];
    const float* bb   = (const float*)d_in[7];
    const float* W1   = (const float*)d_in[8];
    const float* b1   = (const float*)d_in[9];
    const float* W2   = (const float*)d_in[10];
    const float* b2   = (const float*)d_in[11];
    const float* Wxd  = (const float*)d_in[12];
    const float* Whd  = (const float*)d_in[13];
    const float* bd   = (const float*)d_in[14];
    const float* Wo   = (const float*)d_in[15];
    const float* bo   = (const float*)d_in[16];
    float* out = (float*)d_out;

    char* base = (char*)d_ws;
    size_t off = 0;
    auto carve = [&](size_t bytes) -> char* {
        char* p = base + off;
        off = (off + bytes + 255) & ~(size_t)255;
        return p;
    };
    __bf16* wxf_p = (__bf16*)carve((size_t)EE * G4 * 2);
    __bf16* whf_p = (__bf16*)carve((size_t)UU * G4 * 2);
    __bf16* wxb_p = (__bf16*)carve((size_t)EE * G4 * 2);
    __bf16* whb_p = (__bf16*)carve((size_t)UU * G4 * 2);
    __bf16* wxd_p = (__bf16*)carve((size_t)(2 * UU) * G4 * 2);
    __bf16* whd_p = (__bf16*)carve((size_t)DD * G4 * 2);
    __bf16* wo_p  = (__bf16*)carve((size_t)DD * VV * 2);
    __bf16* xsp   = (__bf16*)carve((size_t)TT * BB * EE * 2);
    float*  a     = (float*) carve((size_t)BB * TT * 2 * UU * 4);
    float*  aW1   = (float*) carve((size_t)BB * TT * 10 * 4);
    float*  hdecf[2] = { (float*)carve(BB * DD * 4), (float*)carve(BB * DD * 4) };
    float*  cdec[2]  = { (float*)carve(BB * DD * 4), (float*)carve(BB * DD * 4) };
    __bf16* hdecb[2] = { (__bf16*)carve(BB * DD * 2), (__bf16*)carve(BB * DD * 2) };
    __bf16* ctxp  = (__bf16*)carve((size_t)BB * 2 * UU * 2);
    float*  logits = (float*)carve((size_t)BB * VV * 4);

    auto pack = [&](const float* s, __bf16* d, int K, int N) {
        pack_w<<<(K * N + 255) / 256, 256, 0, stream>>>(s, d, K, N);
    };
    pack(Wxf, wxf_p, EE, G4);
    pack(Whf, whf_p, UU, G4);
    pack(Wxb, wxb_p, EE, G4);
    pack(Whb, whb_p, UU, G4);
    pack(Wxd, wxd_p, 2 * UU, G4);
    pack(Whd, whd_p, DD, G4);
    pack(Wo,  wo_p,  DD, VV);

    embed_k<<<(TT * BB * EE + 255) / 256, 256, 0, stream>>>(x, emb, xsp);

    encoder_k<<<2, 512, 0, stream>>>(xsp, wxf_p, whf_p, bf, wxb_p, whb_p, bb, a);

    aW1_k<<<(BB * TT + 255) / 256, 256, 0, stream>>>(a, W1, aW1);

    dec_init<<<(BB * DD + 255) / 256, 256, 0, stream>>>(hdecf[0], cdec[0], hdecb[0]);

    for (int t = 0; t < TT; ++t) {
        int cur = t & 1, nxt = (t + 1) & 1;
        attn_k<<<BB, TT, 0, stream>>>(a, aW1, W1, b1, W2, b2, hdecf[cur], ctxp, t);
        deccell_k<<<1, 512, 0, stream>>>(ctxp, hdecb[cur], cdec[cur],
                                         wxd_p, whd_p, bd,
                                         hdecf[nxt], hdecb[nxt], cdec[nxt]);
        outproj_k<<<(VV / 16) / 8, 256, 0, stream>>>(hdecb[nxt], wo_p, bo, logits);
        softmax_k<<<BB, 256, 0, stream>>>(logits, out, t);
    }
}
